// Attention_68719476736027
// MI455X (gfx1250) — matrix operand order repaired and checked by reading, not compiled, not run on hardware
//
#include <hip/hip_runtime.h>

// ---------------------------------------------------------------------------
// NATTEN-style block for MI455X (gfx1250, wave32).
//   x(2,96,128,128) -> QKV 1x1 (WMMA f16) -> neighborhood attn (WMMA f16)
//                    -> dwconv5 (VALU)    -> proj 1x1 (WMMA f16) -> out f32
// All channel contractions AND the attention contractions (qk^T, attn*v) run
// on v_wmma_f32_16x16x32_f16 with fp32 accumulation; softmax is register-
// resident with __shfl_xor cross-lane reductions.
// ---------------------------------------------------------------------------

#define DIMC   96
#define HEADS  4
#define HDIM   24
#define KW     7
#define IMG    128
#define HWPIX  16384                    // 128*128
#define SCALE_Q 0.20412414523193154f    // 24^-0.5

typedef __attribute__((ext_vector_type(16))) _Float16 v16h;
typedef __attribute__((ext_vector_type(8)))  _Float16 v8h;
typedef __attribute__((ext_vector_type(8)))  float    v8f;
typedef __attribute__((ext_vector_type(4)))  float    v4f;

__device__ __forceinline__ v16h cat8(v8h lo, v8h hi) {
  v16h r;
#pragma unroll
  for (int i = 0; i < 8; ++i) { r[i] = lo[i]; r[8 + i] = hi[i]; }
  return r;
}

__device__ __forceinline__ v8f wmma_f16(v16h a, v16h b, v8f c) {
  // (neg_a, A, neg_b, B, c_mod, C, reuse_a, reuse_b)
  return __builtin_amdgcn_wmma_f32_16x16x32_f16(false, a, false, b, (short)0, c,
                                                false, false);
}

__device__ __forceinline__ int reflect_idx(int i) {
  return (i < 0) ? -i : ((i > IMG - 1) ? 2 * (IMG - 1) - i : i);
}

// ---------------------------------------------------------------------------
// K0: convert weights to f16 once.  wqkv = [QK_w ; V_w] (288 x 96 row-major),
//     wproj = proj_w (96 x 96).
// ---------------------------------------------------------------------------
__global__ __launch_bounds__(256) void cvt_weights_kernel(
    const float* __restrict__ QK_w, const float* __restrict__ V_w,
    const float* __restrict__ proj_w, _Float16* __restrict__ wqkv,
    _Float16* __restrict__ wproj) {
  const int i = blockIdx.x * 256 + threadIdx.x;
  if (i < 288 * 96) {
    const float v = (i < 192 * 96) ? QK_w[i] : V_w[i - 192 * 96];
    wqkv[i] = (_Float16)v;
  }
  if (i < 96 * 96) wproj[i] = (_Float16)proj_w[i];
}

// ---------------------------------------------------------------------------
// K1: QKV GEMM.  qkv[b][n][p] (f16), n: 0..95 Q*scale, 96..191 K, 192..287 V.
// ---------------------------------------------------------------------------
__global__ __launch_bounds__(256) void qkv_gemm_kernel(
    const float* __restrict__ x, const _Float16* __restrict__ wqkv,
    const float* __restrict__ QK_b, const float* __restrict__ V_b,
    _Float16* __restrict__ qkv) {
  __shared__ _Float16 aS[128 * 112] __attribute__((aligned(32)));

  const int tid  = threadIdx.x;
  const int pix0 = blockIdx.x * 128;
  const int b    = pix0 >> 14;
  const int off0 = pix0 & (HWPIX - 1);
  const float* xb = x + (size_t)b * DIMC * HWPIX;

#pragma unroll 4
  for (int idx = tid; idx < 128 * DIMC; idx += 256) {
    const int c = idx >> 7, p = idx & 127;
    aS[p * 112 + c] = (_Float16)xb[c * HWPIX + off0 + p];
  }
  __syncthreads();

  const int lane = tid & 31, wave = tid >> 5;
  const int m = lane & 15, hi = lane >> 4;

  const _Float16* arow = &aS[(wave * 16 + m) * 112];
  const v16h a0 = cat8(*(const v8h*)(arow + 0  + hi * 16),
                       *(const v8h*)(arow + 8  + hi * 16));
  const v16h a1 = cat8(*(const v8h*)(arow + 32 + hi * 16),
                       *(const v8h*)(arow + 40 + hi * 16));
  const v16h a2 = cat8(*(const v8h*)(arow + 64 + hi * 16),
                       *(const v8h*)(arow + 72 + hi * 16));

  const int prow = pix0 + wave * 16 + hi * 8;

  for (int nt = 0; nt < 18; ++nt) {
    const int n = nt * 16 + m;
    const _Float16* wrow = wqkv + n * 96;
    const v16h b0 = *(const v16h*)(wrow + 0  + hi * 16);
    const v16h b1 = *(const v16h*)(wrow + 32 + hi * 16);
    const v16h b2 = *(const v16h*)(wrow + 64 + hi * 16);
    v8f acc = {};
    acc = wmma_f16(a0, b0, acc);
    acc = wmma_f16(a1, b1, acc);
    acc = wmma_f16(a2, b2, acc);

    const float bias = (n < 2 * DIMC) ? QK_b[n] : V_b[n - 2 * DIMC];
    const float scl  = (n < DIMC) ? SCALE_Q : 1.0f;
    v8h o;
#pragma unroll
    for (int r = 0; r < 8; ++r) o[r] = (_Float16)((acc[r] + bias) * scl);
    *(v8h*)(qkv + ((size_t)b * 288 + n) * HWPIX + (prow & (HWPIX - 1))) = o;
  }
}

// ---------------------------------------------------------------------------
// K2: neighborhood attention on WMMA.
// Block = 128 thr / 4 waves = one (b, head, 4-row x 16-col tile).
// Wave = one 16-pixel row segment (fixed py -> all pixels share window rows).
// Candidates: padded 7x24 halo slice, n = ki*24+cc; halo spatial = n + il*24.
//   logits: 11 x wmma (M=16 px, K=32 pad of 24, N=176 pad of 168)
//   out:    12 x wmma (M=16 px, K=192 pad of 176, N=32 pad of 24)
// ---------------------------------------------------------------------------
__global__ __launch_bounds__(128) void natten_wmma_kernel(
    const _Float16* __restrict__ qkv, const float* __restrict__ rpb,
    float* __restrict__ attnbuf) {
  __shared__ _Float16 kS[248 * 32]    __attribute__((aligned(32))); // [sp][d32]
  __shared__ _Float16 vT[24 * 264]    __attribute__((aligned(32))); // [d][sp pad]
  __shared__ _Float16 qS[4][16][32]   __attribute__((aligned(32)));
  __shared__ _Float16 wS[4][16][192]  __attribute__((aligned(32)));
  __shared__ float    rS[169];

  const int tid  = threadIdx.x;
  const int xt   = blockIdx.x & 7;
  const int yt   = blockIdx.x >> 3;
  const int head = blockIdx.y;
  const int b    = blockIdx.z;
  const int x0 = xt * 16, y0 = yt * 4;
  const int r0 = min(max(y0 - 3, 0), IMG - 10);   // 10-row halo base
  const int c0 = min(max(x0 - 3, 0), IMG - 24);   // 24-col halo base

  const _Float16* qbase = qkv + ((size_t)b * 288 + 0   + head * HDIM) * HWPIX;
  const _Float16* kbase = qkv + ((size_t)b * 288 + 96  + head * HDIM) * HWPIX;
  const _Float16* vbase = qkv + ((size_t)b * 288 + 192 + head * HDIM) * HWPIX;

  // --- zero-fill LDS (WMMA propagates NaN: all pad operands must be finite)
  {
    int* z = (int*)kS;
    for (int i = tid; i < 248 * 32 / 2; i += 128) z[i] = 0;
    z = (int*)vT;
    for (int i = tid; i < 24 * 264 / 2; i += 128) z[i] = 0;
    z = (int*)&qS[0][0][0];
    for (int i = tid; i < 4 * 16 * 32 / 2; i += 128) z[i] = 0;
    z = (int*)&wS[0][0][0];
    for (int i = tid; i < 4 * 16 * 192 / 2; i += 128) z[i] = 0;
  }
  __syncthreads();

  // --- stage K (as [spatial][d]) and V (transposed, [d][spatial])
  for (int idx = tid; idx < 24 * 240; idx += 128) {
    const int d = idx / 240, sp = idx - d * 240;
    const int rr = sp / 24, cc = sp - rr * 24;
    const int g = (r0 + rr) * IMG + (c0 + cc);        // coalesced along cc
    kS[sp * 32 + d]  = kbase[d * HWPIX + g];
    vT[d * 264 + sp] = vbase[d * HWPIX + g];
  }
  // --- stage Q ([seg][px][d32], pad d zeroed above)
  for (int idx = tid; idx < 24 * 64; idx += 128) {
    const int d = idx >> 6, r = idx & 63;
    const int seg = r >> 4, px = r & 15;
    qS[seg][px][d] = qbase[d * HWPIX + (y0 + seg) * IMG + (x0 + px)];
  }
  for (int idx = tid; idx < 169; idx += 128) rS[idx] = rpb[head * 169 + idx];
  __syncthreads();

  const int wave = tid >> 5, lane = tid & 31;
  const int m = lane & 15, hi = lane >> 4;
  const int seg = wave;
  const int py  = y0 + seg;
  const int isA = min(max(py - 3, 0), IMG - KW);
  const int il  = isA - r0;                          // 0..3

  // A-fragment = q (K = d, zero padded 24..31)
  const v16h aq = cat8(*(const v8h*)&qS[seg][m][hi * 16],
                       *(const v8h*)&qS[seg][m][8 + hi * 16]);

  // ---- logits: lane owns candidate column n = nt*16+m, 8 pixel rows
  float lg[11][8];
#pragma unroll
  for (int nt = 0; nt < 11; ++nt) {
    const int n = nt * 16 + m;
    const v16h bk = *(const v16h*)&kS[(il * 24 + n) * 32 + hi * 16];
    v8f acc = {};
    acc = wmma_f16(aq, bk, acc);
    const int ki = n / 24, cc = n - ki * 24;
    const int ccA = c0 + cc;
#pragma unroll
    for (int r = 0; r < 8; ++r) {
      const int px  = x0 + hi * 8 + r;
      const int jsA = min(max(px - 3, 0), IMG - KW);
      const bool ok = (ki < KW) && (ccA >= jsA) && (ccA <= jsA + 6);
      lg[nt][r] = ok ? (acc[r] + rS[(6 + isA + ki - py) * 13 + (6 + ccA - px)])
                     : -3.0e38f;
    }
  }

  // ---- softmax: cross-lane reduce over 16 candidate lanes per half-wave
  float inv[8];
#pragma unroll
  for (int r = 0; r < 8; ++r) {
    float mx = lg[0][r];
#pragma unroll
    for (int nt = 1; nt < 11; ++nt) mx = fmaxf(mx, lg[nt][r]);
#pragma unroll
    for (int s = 1; s < 16; s <<= 1) mx = fmaxf(mx, __shfl_xor(mx, s, 32));
    float sm = 0.f;
#pragma unroll
    for (int nt = 0; nt < 11; ++nt) {
      lg[nt][r] = __expf(lg[nt][r] - mx);
      sm += lg[nt][r];
    }
#pragma unroll
    for (int s = 1; s < 16; s <<= 1) sm += __shfl_xor(sm, s, 32);
    inv[r] = 1.0f / sm;
  }

  // ---- scatter normalized weights (f16) into wS[pixel][cand]
#pragma unroll
  for (int nt = 0; nt < 11; ++nt)
#pragma unroll
    for (int r = 0; r < 8; ++r)
      wS[seg][hi * 8 + r][nt * 16 + m] = (_Float16)(lg[nt][r] * inv[r]);
  __syncthreads();

  // ---- out = W(16x192) * V(192x24): 6 K-steps x 2 d-tiles
  float* obase = attnbuf + ((size_t)b * DIMC + head * HDIM) * HWPIX + py * IMG;
#pragma unroll
  for (int dt = 0; dt < 2; ++dt) {
    const int d = dt * 16 + m;                       // output channel (lane col)
    v8f acc = {};
#pragma unroll
    for (int k0i = 0; k0i < 6; ++k0i) {
      const int k0 = k0i * 32;
      const v16h aw = cat8(*(const v8h*)&wS[seg][m][k0 + hi * 16],
                           *(const v8h*)&wS[seg][m][k0 + 8 + hi * 16]);
      v16h bv = {};
      if (d < HDIM) {
        const int base = d * 264 + il * 24 + k0 + hi * 16;
        bv = cat8(*(const v8h*)&vT[base], *(const v8h*)&vT[base + 8]);
      }
      acc = wmma_f16(aw, bv, acc);
    }
    if (d < HDIM) {
      float* dst = obase + (size_t)d * HWPIX + x0 + hi * 8;
      v4f o0 = {acc[0], acc[1], acc[2], acc[3]};
      v4f o1 = {acc[4], acc[5], acc[6], acc[7]};
      *(v4f*)dst       = o0;
      *(v4f*)(dst + 4) = o1;
    }
  }
}

// ---------------------------------------------------------------------------
// K3: depthwise 5x5 with reflect padding, reading V from qkv (ch 192..287).
// ---------------------------------------------------------------------------
__global__ __launch_bounds__(256) void dwconv_kernel(
    const _Float16* __restrict__ qkv, const float* __restrict__ conv_w,
    const float* __restrict__ conv_b, float* __restrict__ convbuf) {
  __shared__ _Float16 tS[20 * 20];
  __shared__ float    wS[25];

  const int tile = blockIdx.x;
  const int c    = blockIdx.y;
  const int b    = blockIdx.z;
  const int h0 = (tile >> 3) * 16, w0 = (tile & 7) * 16;
  const _Float16* vb = qkv + ((size_t)b * 288 + 192 + c) * HWPIX;

  const int tid = threadIdx.x;
  if (tid < 25) wS[tid] = conv_w[c * 25 + tid];
  for (int idx = tid; idx < 400; idx += 256) {
    const int rr = idx / 20, cc = idx - rr * 20;
    tS[idx] = vb[reflect_idx(h0 - 2 + rr) * IMG + reflect_idx(w0 - 2 + cc)];
  }
  __syncthreads();

  const int ty = tid >> 4, tx = tid & 15;
  float acc = conv_b[c];
#pragma unroll
  for (int ki = 0; ki < 5; ++ki)
#pragma unroll
    for (int kj = 0; kj < 5; ++kj)
      acc += wS[ki * 5 + kj] * (float)tS[(ty + ki) * 20 + (tx + kj)];

  convbuf[((size_t)b * DIMC + c) * HWPIX + (h0 + ty) * IMG + (w0 + tx)] = acc;
}

// ---------------------------------------------------------------------------
// K4: projection GEMM on (attn + conv), f32 output.
// ---------------------------------------------------------------------------
__global__ __launch_bounds__(256) void proj_gemm_kernel(
    const float* __restrict__ attnbuf, const float* __restrict__ convbuf,
    const _Float16* __restrict__ wproj, const float* __restrict__ proj_b,
    float* __restrict__ out) {
  __shared__ _Float16 aS[128 * 112] __attribute__((aligned(32)));

  const int tid  = threadIdx.x;
  const int pix0 = blockIdx.x * 128;
  const int b    = pix0 >> 14;
  const int off0 = pix0 & (HWPIX - 1);

#pragma unroll 4
  for (int idx = tid; idx < 128 * DIMC; idx += 256) {
    const int c = idx >> 7, p = idx & 127;
    const size_t gi = ((size_t)b * DIMC + c) * HWPIX + off0 + p;
    aS[p * 112 + c] = (_Float16)(attnbuf[gi] + convbuf[gi]);
  }
  __syncthreads();

  const int lane = tid & 31, wave = tid >> 5;
  const int m = lane & 15, hi = lane >> 4;

  const _Float16* arow = &aS[(wave * 16 + m) * 112];
  const v16h a0 = cat8(*(const v8h*)(arow + 0  + hi * 16),
                       *(const v8h*)(arow + 8  + hi * 16));
  const v16h a1 = cat8(*(const v8h*)(arow + 32 + hi * 16),
                       *(const v8h*)(arow + 40 + hi * 16));
  const v16h a2 = cat8(*(const v8h*)(arow + 64 + hi * 16),
                       *(const v8h*)(arow + 72 + hi * 16));

  const int prow = pix0 + wave * 16 + hi * 8;

  for (int nt = 0; nt < 6; ++nt) {
    const int n = nt * 16 + m;
    const _Float16* wrow = wproj + n * 96;
    const v16h b0 = *(const v16h*)(wrow + 0  + hi * 16);
    const v16h b1 = *(const v16h*)(wrow + 32 + hi * 16);
    const v16h b2 = *(const v16h*)(wrow + 64 + hi * 16);
    v8f acc = {};
    acc = wmma_f16(a0, b0, acc);
    acc = wmma_f16(a1, b1, acc);
    acc = wmma_f16(a2, b2, acc);

    const float bias = proj_b[n];
    float* dst = out + ((size_t)b * DIMC + n) * HWPIX + (prow & (HWPIX - 1));
    v4f o0 = {acc[0] + bias, acc[1] + bias, acc[2] + bias, acc[3] + bias};
    v4f o1 = {acc[4] + bias, acc[5] + bias, acc[6] + bias, acc[7] + bias};
    *(v4f*)dst       = o0;
    *(v4f*)(dst + 4) = o1;
  }
}

// ---------------------------------------------------------------------------
extern "C" void kernel_launch(void* const* d_in, const int* in_sizes, int n_in,
                              void* d_out, int out_size, void* d_ws, size_t ws_size,
                              hipStream_t stream) {
  (void)in_sizes; (void)n_in; (void)out_size; (void)ws_size;

  const float* x      = (const float*)d_in[0];
  const float* V_w    = (const float*)d_in[1];
  const float* V_b    = (const float*)d_in[2];
  const float* QK_w   = (const float*)d_in[3];
  const float* QK_b   = (const float*)d_in[4];
  const float* conv_w = (const float*)d_in[5];
  const float* conv_b = (const float*)d_in[6];
  const float* proj_w = (const float*)d_in[7];
  const float* proj_b = (const float*)d_in[8];
  const float* rpb    = (const float*)d_in[9];

  char* ws = (char*)d_ws;
  _Float16* wqkv  = (_Float16*)(ws);                       //    55,296 B
  _Float16* wproj = (_Float16*)(ws + 55296);               //    18,432 B
  _Float16* qkv   = (_Float16*)(ws + 73728);               // 18,874,368 B
  float* attnbuf  = (float*)(ws + 73728 + 18874368);       // 12,582,912 B
  float* convbuf  = (float*)(ws + 73728 + 18874368 + 12582912);

  cvt_weights_kernel<<<108, 256, 0, stream>>>(QK_w, V_w, proj_w, wqkv, wproj);
  qkv_gemm_kernel<<<256, 256, 0, stream>>>(x, wqkv, QK_b, V_b, qkv);
  natten_wmma_kernel<<<dim3(256, HEADS, 2), 128, 0, stream>>>(qkv, rpb, attnbuf);
  dwconv_kernel<<<dim3(64, DIMC, 2), 256, 0, stream>>>(qkv, conv_w, conv_b, convbuf);
  proj_gemm_kernel<<<256, 256, 0, stream>>>(attnbuf, convbuf, wproj, proj_b,
                                            (float*)d_out);
}
